// Adapter_884763263188
// MI455X (gfx1250) — matrix operand order repaired and checked by reading, not compiled, not run on hardware
//
#include <hip/hip_runtime.h>
#include <hip/hip_bf16.h>
#include <math.h>

typedef __bf16 bf16_t;
typedef __attribute__((ext_vector_type(16))) __bf16 v16bf;
typedef __attribute__((ext_vector_type(8)))  __bf16 v8bf;
typedef __attribute__((ext_vector_type(8)))  float  v8f;
typedef __attribute__((ext_vector_type(4)))  int    i32x4;
typedef __attribute__((address_space(1))) i32x4 as1_i32x4;
typedef __attribute__((address_space(3))) i32x4 as3_i32x4;

#define TS 1024
#define TB 8
#define TD 512
#define TV 8000
#define TE 1024
#define TM (TS*TB)

#define BM 128
#define BN 128
#define BK 32
#define LSTR 40   // LDS row stride in bf16 elements (80B, 16B-aligned rows)

// ---------------- CDNA5 async global->LDS copy (ASYNCcnt domain) ----------------
#if defined(__has_builtin)
#  if __has_builtin(__builtin_amdgcn_global_load_async_to_lds_b128)
#    define HAVE_ASYNC_B128 1
#  endif
#  if __has_builtin(__builtin_amdgcn_s_wait_asynccnt)
#    define HAVE_WAIT_ASYNC 1
#  endif
#endif

__device__ __forceinline__ void async_cp16(const void* g, void* l) {
#ifdef HAVE_ASYNC_B128
  __builtin_amdgcn_global_load_async_to_lds_b128(
      (as1_i32x4*)g, (as3_i32x4*)l, 0, 0);
#else
  // VDST = per-lane LDS byte offset, VADDR = 64-bit global address
  unsigned loff = (unsigned)(size_t)(__attribute__((address_space(3))) void*)l;
  asm volatile("global_load_async_to_lds_b128 %0, %1, off"
               :: "v"(loff), "v"(g) : "memory");
#endif
}

__device__ __forceinline__ void wait_async0() {
#ifdef HAVE_WAIT_ASYNC
  __builtin_amdgcn_s_wait_asynccnt(0);
#else
  asm volatile("s_wait_asynccnt 0x0" ::: "memory");
#endif
}

// ---------------- K2a: f32 -> bf16 ----------------
__global__ __launch_bounds__(256) void cvt_f32_bf16(const float* __restrict__ src,
                                                    bf16_t* __restrict__ dst, long n) {
  long i = (long)blockIdx.x * blockDim.x + threadIdx.x;
  long stride = (long)gridDim.x * blockDim.x;
  for (; i < n; i += stride) dst[i] = (bf16_t)src[i];
}

// ---------------- K2b: emb [V][D] f32 -> embT [D][V] bf16 ----------------
__global__ __launch_bounds__(256) void transpose_emb_kernel(const float* __restrict__ src,
                                                            bf16_t* __restrict__ dst) {
  long n = (long)TD * TV;
  long i = (long)blockIdx.x * blockDim.x + threadIdx.x;
  long stride = (long)gridDim.x * blockDim.x;
  for (; i < n; i += stride) {
    long d = i / TV, v = i - d * TV;
    dst[i] = (bf16_t)src[v * (long)TD + d];
  }
}

// ---------------- K1: fused online softmax + argmax per row ----------------
// Writes UNNORMALIZED exp(x - rowmax) as bf16; recip row-sum folded into GEMM3.
__global__ __launch_bounds__(256) void softmax_pred_kernel(const float* __restrict__ logit,
                                                           bf16_t* __restrict__ dist,
                                                           float* __restrict__ rsum,
                                                           int* __restrict__ pred) {
  const int row = blockIdx.x;                  // row = s*B + b
  const float* x = logit + (size_t)row * TV;
  const int tid = threadIdx.x;

  float m = -INFINITY, s = 0.f;
  float bv = -INFINITY; int bi = 0x7fffffff;
  for (int v = tid; v < TV; v += 256) {
    float xv = x[v];
    if (xv > bv) { bv = xv; bi = v; }
    if (xv > m)  { s = s * __expf(m - xv) + 1.f; m = xv; }
    else         { s += __expf(xv - m); }
  }

  __shared__ float sm[256], ss[256], sbv[256];
  __shared__ int   sbi[256];
  sm[tid] = m; ss[tid] = s; sbv[tid] = bv; sbi[tid] = bi;
  for (int off = 128; off > 0; off >>= 1) {
    __syncthreads();
    if (tid < off) {
      float m1 = sm[tid], s1 = ss[tid];
      float m2 = sm[tid + off], s2 = ss[tid + off];
      float Mx = fmaxf(m1, m2);
      ss[tid] = s1 * __expf(m1 - Mx) + s2 * __expf(m2 - Mx);
      sm[tid] = Mx;
      float v2 = sbv[tid + off]; int i2 = sbi[tid + off];
      if (v2 > sbv[tid] || (v2 == sbv[tid] && i2 < sbi[tid])) { sbv[tid] = v2; sbi[tid] = i2; }
    }
  }
  __syncthreads();
  const float rowmax = sm[0];
  if (tid == 0) { pred[row] = sbi[0]; rsum[row] = 1.f / ss[0]; }

  bf16_t* drow = dist + (size_t)row * TV;
  for (int v = tid; v < TV; v += 256)
    drow[v] = (bf16_t)__expf(x[v] - rowmax);
}

// ---------------- K3: bf16 WMMA GEMM  C[M,N] = A[M,K] * B[N,K]^T ----------------
// 256 threads = 8 waves (4 in M, 2 in N). Wave tile = 32x64 (2x4 WMMA tiles).
// Block tile 128x128, double-buffered LDS staged via ASYNC global->LDS copies.
template <bool RELU, bool OUT_BF16>
__global__ __launch_bounds__(256) void gemm_bf16_kernel(
    const bf16_t* __restrict__ A, const bf16_t* __restrict__ Bm,
    const float* __restrict__ bias, const float* __restrict__ rowscale,
    void* __restrict__ Cout, int M, int N, int K) {
  __shared__ __align__(16) bf16_t sA[2][BM * LSTR];
  __shared__ __align__(16) bf16_t sB[2][BN * LSTR];

  const int tid  = threadIdx.x;
  const int lane = tid & 31;
  const int wave = tid >> 5;
  const int wm   = wave >> 1;        // 0..3
  const int wn   = wave & 1;         // 0..1
  const int half = lane >> 4;        // lane half selects K-group per ISA layout
  const int l16  = lane & 15;

  const int rowBase = blockIdx.y * BM;
  const int colBase = blockIdx.x * BN;

  v8f acc[2][4];
#pragma unroll
  for (int mi = 0; mi < 2; ++mi)
#pragma unroll
    for (int ni = 0; ni < 4; ++ni) { v8f z = {}; acc[mi][ni] = z; }

  const int numTiles = K / BK;

  auto loadTileAsync = [&](int kt, int buf) {
    const long k0 = (long)kt * BK;
#pragma unroll
    for (int i = 0; i < 2; ++i) {
      int c  = tid + i * 256;        // 0..511
      int r  = c >> 2;               // 0..127
      int cc = c & 3;                // 16B chunk within 64B row
      async_cp16(A  + (size_t)(rowBase + r) * K + k0 + cc * 8,
                 &sA[buf][r * LSTR + cc * 8]);
      async_cp16(Bm + (size_t)(colBase + r) * K + k0 + cc * 8,
                 &sB[buf][r * LSTR + cc * 8]);
    }
  };

  loadTileAsync(0, 0);
  wait_async0();
  __syncthreads();

  for (int kt = 0; kt < numTiles; ++kt) {
    const int cur = kt & 1;
    if (kt + 1 < numTiles) loadTileAsync(kt + 1, cur ^ 1);

    // A fragments: lane<16 holds K 0..7 & 16..23 of row M=l16; lane>=16 holds K 8..15 & 24..31
    v16bf afrag[2];
#pragma unroll
    for (int mi = 0; mi < 2; ++mi) {
      const bf16_t* p = &sA[cur][(wm * 32 + mi * 16 + l16) * LSTR + half * 8];
      union { v16bf v; v8bf h[2]; } u;
      u.h[0] = *(const v8bf*)p;
      u.h[1] = *(const v8bf*)(p + 16);
      afrag[mi] = u.v;
    }
    // B fragments: lane<16 holds K 0..7 & 16..23 of col N=l16; lane>=16 holds K 8..15 & 24..31
    v16bf bfrag[4];
#pragma unroll
    for (int ni = 0; ni < 4; ++ni) {
      const bf16_t* p = &sB[cur][(wn * 64 + ni * 16 + l16) * LSTR + half * 8];
      union { v16bf v; v8bf h[2]; } u;
      u.h[0] = *(const v8bf*)p;
      u.h[1] = *(const v8bf*)(p + 16);
      bfrag[ni] = u.v;
    }
#pragma unroll
    for (int mi = 0; mi < 2; ++mi)
#pragma unroll
      for (int ni = 0; ni < 4; ++ni)
        acc[mi][ni] = __builtin_amdgcn_wmma_f32_16x16x32_bf16(
            false, afrag[mi], false, bfrag[ni], (short)0, acc[mi][ni], false, false);

    // async loads for the next stage must have landed in LDS before anyone reads it
    wait_async0();
    __syncthreads();
  }

  // Epilogue per ISA C/D layout: VGPR i -> M = i + half*8, N = l16
#pragma unroll
  for (int mi = 0; mi < 2; ++mi) {
#pragma unroll
    for (int ni = 0; ni < 4; ++ni) {
#pragma unroll
      for (int i = 0; i < 8; ++i) {
        const int grow = rowBase + wm * 32 + mi * 16 + half * 8 + i;
        const int gcol = colBase + wn * 64 + ni * 16 + l16;
        float v = acc[mi][ni][i];
        if (rowscale) v *= rowscale[grow];
        if (bias)     v += bias[gcol];
        if (RELU)     v = fmaxf(v, 0.f);
        if (OUT_BF16) ((bf16_t*)Cout)[(size_t)grow * N + gcol] = (bf16_t)v;
        else          ((float*)Cout)[(size_t)grow * N + gcol]  = v;
      }
    }
  }
}

// ---------------- K4: dual LayerNorm + add (D=512, 256 threads, 2 elems/thread) ----
__global__ __launch_bounds__(256) void ln_combine_kernel(
    const float* __restrict__ lin, const float* __restrict__ soft,
    const float* __restrict__ g1, const float* __restrict__ b1v,
    const float* __restrict__ g2, const float* __restrict__ b2v,
    float* __restrict__ outp) {
  const int row = blockIdx.x;
  const int tid = threadIdx.x;
  const float* lr = lin  + (size_t)row * TD;
  const float* sr = soft + (size_t)row * TD;
  float l0 = lr[tid], l1 = lr[tid + 256];
  float s0 = sr[tid], s1 = sr[tid + 256];

  __shared__ float red[4][256];
  red[0][tid] = l0 + l1;
  red[1][tid] = l0 * l0 + l1 * l1;
  red[2][tid] = s0 + s1;
  red[3][tid] = s0 * s0 + s1 * s1;
  for (int off = 128; off > 0; off >>= 1) {
    __syncthreads();
    if (tid < off) {
#pragma unroll
      for (int j = 0; j < 4; ++j) red[j][tid] += red[j][tid + off];
    }
  }
  __syncthreads();
  const float invD = 1.f / TD;
  const float m1 = red[0][0] * invD, v1 = red[1][0] * invD - m1 * m1;
  const float m2 = red[2][0] * invD, v2 = red[3][0] * invD - m2 * m2;
  const float r1 = rsqrtf(v1 + 1e-5f), r2 = rsqrtf(v2 + 1e-5f);

  float* o = outp + (size_t)row * TD;
  o[tid]       = (l0 - m1) * r1 * g1[tid]       + b1v[tid]
               + (s0 - m2) * r2 * g2[tid]       + b2v[tid];
  o[tid + 256] = (l1 - m1) * r1 * g1[tid + 256] + b1v[tid + 256]
               + (s1 - m2) * r2 * g2[tid + 256] + b2v[tid + 256];
}

// ---------------- K5: per-batch run-length segmentation ----------------
__global__ void segment_kernel(const int* __restrict__ pred,
                               const unsigned char* __restrict__ padding,
                               int* __restrict__ runstart, int* __restrict__ runcnt,
                               int* __restrict__ nseg) {
  const int b = blockIdx.x;
  if (threadIdx.x == 0) {
    int prev = -1, n = 0;
    for (int s = 0; s < TS; ++s) {
      const int p = pred[s * TB + b];
      const bool valid = (padding[b * TS + s] == 0);
      if (valid && p != prev) { runstart[b * TS + n] = s; runcnt[b * TS + n] = 0; ++n; }
      if (valid && n > 0) runcnt[b * TS + n - 1]++;
      prev = p;
    }
    nseg[b] = n;
  }
}

// ---------------- K6: deterministic gather/average compress + padding flags ------
__global__ __launch_bounds__(256) void compress_kernel(
    const float* __restrict__ outf, const int* __restrict__ runstart,
    const int* __restrict__ runcnt, const int* __restrict__ nseg,
    float* __restrict__ d_out) {
  const int t = blockIdx.x;      // segment (padded to S)
  const int b = blockIdx.y;
  const int tid = threadIdx.x;
  const int ns = nseg[b];
  float* dst = d_out + ((size_t)t * TB + b) * TD;

  if (t >= ns) {
    for (int d = tid; d < TD; d += 256) dst[d] = 0.f;
  } else {
    const int start = runstart[b * TS + t];
    const int c = runcnt[b * TS + t];
    const float inv = 1.f / (float)c;
    for (int d = tid; d < TD; d += 256) {
      float acc = 0.f;
      for (int f = 0; f < c; ++f)
        acc += outf[((size_t)(start + f) * TB + b) * TD + d];
      dst[d] = acc * inv;
    }
  }
  if (tid == 0)
    d_out[(size_t)TS * TB * TD + (size_t)b * TS + t] = (t >= ns) ? 1.f : 0.f;
}

// ---------------- host side ----------------
extern "C" void kernel_launch(void* const* d_in, const int* in_sizes, int n_in,
                              void* d_out, int out_size, void* d_ws, size_t ws_size,
                              hipStream_t stream) {
  const float* rep    = (const float*)d_in[0];
  const float* logit  = (const float*)d_in[1];
  const unsigned char* padding = (const unsigned char*)d_in[2];
  const float* w1     = (const float*)d_in[3];
  const float* b1     = (const float*)d_in[4];
  const float* w2     = (const float*)d_in[5];
  const float* b2     = (const float*)d_in[6];
  const float* ln1_g  = (const float*)d_in[7];
  const float* ln1_b  = (const float*)d_in[8];
  const float* emb    = (const float*)d_in[9];
  const float* eln_g  = (const float*)d_in[10];
  const float* eln_b  = (const float*)d_in[11];
  float* outp = (float*)d_out;

  char* ws = (char*)d_ws;
  size_t off = 0;
  auto carve = [&](size_t bytes) -> char* {
    char* p = ws + off;
    off = (off + bytes + 255) & ~(size_t)255;
    return p;
  };
  bf16_t* dist_bf  = (bf16_t*)carve((size_t)TM * TV * 2);   // 131 MB
  bf16_t* rep_bf   = (bf16_t*)carve((size_t)TM * TD * 2);
  bf16_t* w1_bf    = (bf16_t*)carve((size_t)TE * TD * 2);
  bf16_t* w2_bf    = (bf16_t*)carve((size_t)TD * TE * 2);
  bf16_t* embT_bf  = (bf16_t*)carve((size_t)TD * TV * 2);
  bf16_t* h_bf     = (bf16_t*)carve((size_t)TM * TE * 2);
  float*  lin_f    = (float*) carve((size_t)TM * TD * 4);
  float*  soft_f   = (float*) carve((size_t)TM * TD * 4);
  float*  out_f    = (float*) carve((size_t)TM * TD * 4);
  float*  rsum     = (float*) carve((size_t)TM * 4);
  int*    pred     = (int*)   carve((size_t)TM * 4);
  int*    runstart = (int*)   carve((size_t)TB * TS * 4);
  int*    runcnt   = (int*)   carve((size_t)TB * TS * 4);
  int*    nseg     = (int*)   carve((size_t)TB * 4);

  // K2: conversions / transpose (independent of K1)
  cvt_f32_bf16<<<2048, 256, 0, stream>>>(rep, rep_bf, (long)TM * TD);
  cvt_f32_bf16<<<512,  256, 0, stream>>>(w1,  w1_bf,  (long)TE * TD);
  cvt_f32_bf16<<<512,  256, 0, stream>>>(w2,  w2_bf,  (long)TD * TE);
  transpose_emb_kernel<<<4096, 256, 0, stream>>>(emb, embT_bf);

  // K1: softmax + argmax (dominant bandwidth: 2 passes over 262MB logits)
  softmax_pred_kernel<<<TM, 256, 0, stream>>>(logit, dist_bf, rsum, pred);

  // K3a: h = relu(rep @ w1^T + b1)          [8192 x 1024], K=512
  gemm_bf16_kernel<true, true><<<dim3(TE / BN, TM / BM), 256, 0, stream>>>(
      rep_bf, w1_bf, b1, nullptr, (void*)h_bf, TM, TE, TD);

  // K3b: lin = h @ w2^T + b2                [8192 x 512],  K=1024
  gemm_bf16_kernel<false, false><<<dim3(TD / BN, TM / BM), 256, 0, stream>>>(
      h_bf, w2_bf, b2, nullptr, (void*)lin_f, TM, TD, TE);

  // K3c: soft = (exp_unnorm @ embT^T) * rsum [8192 x 512], K=8000 (the big one)
  gemm_bf16_kernel<false, false><<<dim3(TD / BN, TM / BM), 256, 0, stream>>>(
      dist_bf, embT_bf, nullptr, rsum, (void*)soft_f, TM, TD, TV);

  // K4: out = LN(lin)*g+b + LN(soft)*g+b
  ln_combine_kernel<<<TM, 256, 0, stream>>>(lin_f, soft_f, ln1_g, ln1_b, eln_g, eln_b, out_f);

  // K5: segmentation per batch
  segment_kernel<<<TB, 32, 0, stream>>>(pred, padding, runstart, runcnt, nseg);

  // K6: compress + padding flags (writes every output element exactly once)
  compress_kernel<<<dim3(TS, TB), 256, 0, stream>>>(out_f, runstart, runcnt, nseg, outp);
}
